// GPS_30846455120747
// MI455X (gfx1250) — compile-verified, matrix operand order checked by reading, not run on hardware
//
#include <hip/hip_runtime.h>
#include <hip/hip_bf16.h>

// ============================================================================
// GraphGPS (2 layers) for MI455X / gfx1250 (wave32, WMMA f32_16x16x32_f16).
// All dense matmuls + attention use v_wmma with f32 accumulation; softmax,
// batchnorm, residuals, log-softmax stay f32. Attention is flash-style
// (online softmax), so no N x N score matrix is materialized.
// Tile staging uses the CDNA5 async global->LDS copy path (ASYNCcnt) where
// the copy is unpredicated; predicated/transposing stages use the VGPR path.
// ============================================================================

typedef __attribute__((ext_vector_type(8)))  _Float16 v8h;
typedef __attribute__((ext_vector_type(16))) _Float16 v16h;
typedef __attribute__((ext_vector_type(8)))  float    v8f;

#define N_NODES 8192
#define E_EDGES 262144
#define D_CH    128
#define N_CLS   40

static __device__ __forceinline__ v8f vzero8() {
  v8f z;
#pragma unroll
  for (int i = 0; i < 8; ++i) z[i] = 0.0f;
  return z;
}

// Async global -> LDS copy of 16B per lane (GLOBAL_LOAD_ASYNC_TO_LDS_B128,
// tracked by ASYNCcnt). LDS destination = low 32 bits of the generic shared
// pointer (flat->LDS truncation, ISA 10.2).
static __device__ __forceinline__ void async_g2l_b128(_Float16* lds_dst,
                                                      const _Float16* g_src) {
  unsigned loff = (unsigned)(unsigned long long)(void*)lds_dst;
  asm volatile("global_load_async_to_lds_b128 %0, %1, off"
               :: "v"(loff), "v"(g_src)
               : "memory");
}
static __device__ __forceinline__ void wait_async0() {
  asm volatile("s_wait_asynccnt 0x0" ::: "memory");
}

// Load one 16-half A/B WMMA fragment from an LDS row.
// Row layout (16-bit A/B, 16x16x32): lane[3:0] = M (or N) row, lane[4] = khalf.
// elements 0..7  <- K = khalf*8 + 0..7      (contiguous 16B)
// elements 8..15 <- K = 16 + khalf*8 + 0..7 (contiguous 16B)
// p must point at &row[khalf*8].
static __device__ __forceinline__ v16h ldfrag(const _Float16* p) {
  union { v16h v; v8h h[2]; } u;
  u.h[0] = *(const v8h*)(p);
  u.h[1] = *(const v8h*)(p + 16);
  return u.v;
}

// ----------------------------------------------------------------------------
// GEMM: C[M x Nc] = A[M x K] @ B[K x Nc] (+bias, opt relu)
// A   : f16 row-major [M x K]
// BT  : f16 row-major [Nc x K]  (B transposed)
// Cf  : f32 out (nullable), Ch : f16 out (nullable)
// Block: 256 threads (8 waves), tile 128(M) x 64(N), K-step 32.
// A tile staged via async global->LDS; B^T tile via VGPRs (zero padding).
// ----------------------------------------------------------------------------
__global__ __launch_bounds__(256) void k_gemm_f16(
    const _Float16* __restrict__ A, const _Float16* __restrict__ BT,
    const float* __restrict__ bias, float* __restrict__ Cf,
    _Float16* __restrict__ Ch, int M, int K, int Ncols, int relu)
{
  constexpr int LDA = 48;  // halves per LDS row (96B, 16B aligned)
  __shared__ _Float16 As[128 * LDA];
  __shared__ _Float16 Bs[64 * LDA];
  const int tid  = threadIdx.x;
  const int row0 = blockIdx.x * 128;
  const int col0 = blockIdx.y * 64;
  const int w    = tid >> 5, lane = tid & 31;
  const int wm   = (w & 3) * 32;   // wave M offset in block tile
  const int wn   = (w >> 2) * 32;  // wave N offset in block tile
  const int kh   = lane >> 4, lr = lane & 15;

  v8f acc[2][2];
  acc[0][0] = vzero8(); acc[0][1] = vzero8();
  acc[1][0] = vzero8(); acc[1][1] = vzero8();

  for (int k0 = 0; k0 < K; k0 += 32) {
    {  // stage A tile: 128 rows x 32 halves (async -> LDS)
      int r = tid >> 1, cb = (tid & 1) * 16;
      const _Float16* g = A + (size_t)(row0 + r) * K + k0 + cb;
      _Float16* l = &As[r * LDA + cb];
      async_g2l_b128(l, g);
      async_g2l_b128(l + 8, g + 8);
    }
    {  // stage B^T tile: 64 rows x 32 halves (zero-pad past Ncols)
      int r = tid >> 2, cb = (tid & 3) * 8;
      v8h bval;
#pragma unroll
      for (int i = 0; i < 8; ++i) bval[i] = (_Float16)0.0f;
      if (col0 + r < Ncols)
        bval = *(const v8h*)(BT + (size_t)(col0 + r) * K + k0 + cb);
      *(v8h*)&Bs[r * LDA + cb] = bval;
    }
    wait_async0();
    __syncthreads();
    v16h b0 = ldfrag(&Bs[(wn +  0 + lr) * LDA + kh * 8]);
    v16h b1 = ldfrag(&Bs[(wn + 16 + lr) * LDA + kh * 8]);
#pragma unroll
    for (int ti = 0; ti < 2; ++ti) {
      v16h a = ldfrag(&As[(wm + ti * 16 + lr) * LDA + kh * 8]);
      acc[ti][0] = __builtin_amdgcn_wmma_f32_16x16x32_f16(
          false, a, false, b0, (short)0, acc[ti][0], false, false);
      acc[ti][1] = __builtin_amdgcn_wmma_f32_16x16x32_f16(
          false, a, false, b1, (short)0, acc[ti][1], false, false);
    }
    __syncthreads();
  }
  // C layout: VGPR r -> M = r + kh*8 ; N = lr
#pragma unroll
  for (int ti = 0; ti < 2; ++ti)
#pragma unroll
    for (int tj = 0; tj < 2; ++tj)
#pragma unroll
      for (int r = 0; r < 8; ++r) {
        int m = row0 + wm + ti * 16 + kh * 8 + r;
        int n = col0 + wn + tj * 16 + lr;
        if (n < Ncols) {
          float v = acc[ti][tj][r];
          if (bias) v += bias[n];
          if (relu) v = fmaxf(v, 0.0f);
          size_t o = (size_t)m * Ncols + n;
          if (Cf) Cf[o] = v;
          if (Ch) Ch[o] = (_Float16)v;
        }
      }
}

// ----------------------------------------------------------------------------
// Flash attention: O = softmax(Q K^T / sqrt(d)) V,  d = 128, single head.
// Block: 64 query rows, 256 threads (8 waves); loops over keys in tiles of 32.
// O accumulator lives in registers (16 rows x 64 cols per wave).
// Q and K tiles staged async -> LDS; V^T transposed through VGPRs.
// ----------------------------------------------------------------------------
__global__ __launch_bounds__(256) void k_flash_attn(
    const _Float16* __restrict__ Qh, const _Float16* __restrict__ Kh,
    const _Float16* __restrict__ Vh, float* __restrict__ Of,
    _Float16* __restrict__ Oh, int N)
{
  constexpr int LQ = 136;  // halves/row for Q,K tiles (272B, 16B aligned)
  constexpr int LV = 40;   // halves/row for V^T and P tiles (80B aligned)
  constexpr int LS = 36;   // floats/row for S tile
  __shared__ _Float16 Qs[64 * LQ];
  __shared__ _Float16 Ks[32 * LQ];
  __shared__ _Float16 VTs[128 * LV];
  __shared__ float    Ss[64 * LS];
  __shared__ _Float16 Ps[64 * LV];
  __shared__ float    mrow[64], lrow[64], arow[64];

  const int tid  = threadIdx.x;
  const int row0 = blockIdx.x * 64;
  const int w    = tid >> 5, lane = tid & 31;
  const int kh   = lane >> 4, lr = lane & 15;
  const int sm0  = (w & 3) * 16;   // query-row tile for this wave
  const int sn0  = (w >> 2) * 16;  // S column tile (0 or 16)
  const int on0  = (w >> 2) * 64;  // O column base  (0 or 64)
  const float scale = 0.08838834764831845f;  // 1/sqrt(128)

  {  // stage Q tile once: 64 x 128 halves (async -> LDS)
    int r = tid >> 2, cb = (tid & 3) * 32;
    const _Float16* g = Qh + (size_t)(row0 + r) * 128 + cb;
    _Float16* l = &Qs[r * LQ + cb];
#pragma unroll
    for (int i = 0; i < 4; ++i)
      async_g2l_b128(l + i * 8, g + i * 8);
  }
  if (tid < 64) { mrow[tid] = -3.0e38f; lrow[tid] = 0.0f; }

  v8f oacc[4];
#pragma unroll
  for (int t = 0; t < 4; ++t) oacc[t] = vzero8();

  for (int kb = 0; kb < N; kb += 32) {
    if (kb + 32 < N) {  // hint next key/value tiles toward the caches
      __builtin_prefetch(Kh + (size_t)(kb + 32 + lane) * 128, 0, 0);
      __builtin_prefetch(Vh + (size_t)(kb + 32 + lane) * 128, 0, 0);
    }
    {  // stage K tile (async, row-major) and V^T tile (VGPR transpose)
      int r = tid >> 3, cb = (tid & 7) * 16;
      const _Float16* gk = Kh + (size_t)(kb + r) * 128 + cb;
      _Float16* lk = &Ks[r * LQ + cb];
      async_g2l_b128(lk, gk);
      async_g2l_b128(lk + 8, gk + 8);
      const _Float16* gv = Vh + (size_t)(kb + r) * 128 + cb;
      v8h v0 = *(const v8h*)gv;
      v8h v1 = *(const v8h*)(gv + 8);
#pragma unroll
      for (int i = 0; i < 8; ++i) {
        VTs[(cb + i)     * LV + r] = v0[i];
        VTs[(cb + 8 + i) * LV + r] = v1[i];
      }
    }
    wait_async0();  // Q tile (first iter) + K tile complete in LDS
    __syncthreads();
    // S tile = Q K^T (one 16x16 tile per wave, 4 K-steps of 32)
    v8f sacc = vzero8();
#pragma unroll
    for (int k0 = 0; k0 < 128; k0 += 32) {
      v16h a = ldfrag(&Qs[(sm0 + lr) * LQ + k0 + kh * 8]);
      v16h b = ldfrag(&Ks[(sn0 + lr) * LQ + k0 + kh * 8]);
      sacc = __builtin_amdgcn_wmma_f32_16x16x32_f16(
          false, a, false, b, (short)0, sacc, false, false);
    }
#pragma unroll
    for (int r = 0; r < 8; ++r)
      Ss[(sm0 + kh * 8 + r) * LS + sn0 + lr] = sacc[r] * scale;
    __syncthreads();
    // online softmax: one thread per query row
    if (tid < 64) {
      const int r = tid;
      float sv[32];
      float mold = mrow[r];
      float mx = mold;
#pragma unroll
      for (int c = 0; c < 32; ++c) { sv[c] = Ss[r * LS + c]; mx = fmaxf(mx, sv[c]); }
      float alpha = __expf(mold - mx);
      float lsum = 0.0f;
#pragma unroll
      for (int c = 0; c < 32; ++c) {
        float p = __expf(sv[c] - mx);
        Ps[r * LV + c] = (_Float16)p;
        lsum += p;
      }
      lrow[r] = lrow[r] * alpha + lsum;
      mrow[r] = mx;
      arow[r] = alpha;
    }
    __syncthreads();
    // rescale O accumulator, then O += P @ V
    float av[8];
#pragma unroll
    for (int r = 0; r < 8; ++r) av[r] = arow[sm0 + kh * 8 + r];
#pragma unroll
    for (int t = 0; t < 4; ++t)
#pragma unroll
      for (int r = 0; r < 8; ++r) oacc[t][r] *= av[r];
    v16h pa = ldfrag(&Ps[(sm0 + lr) * LV + kh * 8]);
#pragma unroll
    for (int t = 0; t < 4; ++t) {
      v16h b = ldfrag(&VTs[(on0 + t * 16 + lr) * LV + kh * 8]);
      oacc[t] = __builtin_amdgcn_wmma_f32_16x16x32_f16(
          false, pa, false, b, (short)0, oacc[t], false, false);
    }
    __syncthreads();
  }
  float lv[8];
#pragma unroll
  for (int r = 0; r < 8; ++r) lv[r] = 1.0f / lrow[sm0 + kh * 8 + r];
#pragma unroll
  for (int t = 0; t < 4; ++t)
#pragma unroll
    for (int r = 0; r < 8; ++r) {
      int m = row0 + sm0 + kh * 8 + r;
      int n = on0 + t * 16 + lr;
      float v = oacc[t][r] * lv[r];
      size_t o = (size_t)m * 128 + n;
      if (Of) Of[o] = v;
      Oh[o] = (_Float16)v;
    }
}

// ----------------------------------------------------------------------------
// Scalar / elementwise kernels
// ----------------------------------------------------------------------------
__global__ void k_fill(float* p, float v, int n) {
  int i = blockIdx.x * 256 + threadIdx.x;
  if (i < n) p[i] = v;
}
__global__ void k_deg(const int* __restrict__ dst, float* deg, int e) {
  int i = blockIdx.x * 256 + threadIdx.x;
  if (i < e) atomicAdd(&deg[dst[i]], 1.0f);
}
__global__ void k_rsqrt(const float* __restrict__ deg, float* dinv, int n) {
  int i = blockIdx.x * 256 + threadIdx.x;
  if (i < n) dinv[i] = rsqrtf(deg[i]);
}
__global__ void k_f32_to_f16(const float* __restrict__ a, _Float16* o, size_t n) {
  size_t i = (size_t)blockIdx.x * 256 + threadIdx.x;
  if (i < n) o[i] = (_Float16)a[i];
}
// WT[n*K + k] = W[k*Nc + n]  (f32 -> f16 transpose)
__global__ void k_transpose_f16(const float* __restrict__ W, _Float16* WT, int K, int Nc) {
  int i = blockIdx.x * 256 + threadIdx.x;
  if (i < K * Nc) {
    int k = i / Nc, n = i % Nc;
    WT[(size_t)n * K + k] = (_Float16)W[i];
  }
}
// t[i,f] = h[i,f] * dinv[i]^2   (self-loop term of sym-normalized GCN)
__global__ void k_gcn_self(const float* __restrict__ hg, const float* __restrict__ dinv,
                           float* t, size_t nc, int C) {
  size_t i = (size_t)blockIdx.x * 256 + threadIdx.x;
  if (i < nc) {
    int node = (int)(i / (size_t)C);
    float di = dinv[node];
    t[i] = hg[i] * di * di;
  }
}
// t[dst] += h[src] * dinv[src]*dinv[dst], 32 threads/edge * 4 feats/thread
__global__ void k_gcn_scatter(const int* __restrict__ src, const int* __restrict__ dst,
                              const float* __restrict__ hg, const float* __restrict__ dinv,
                              float* t, int E) {
  size_t idx = (size_t)blockIdx.x * 256 + threadIdx.x;
  int e = (int)(idx >> 5);
  if (e >= E) return;
  int f = (int)(idx & 31) * 4;
  int s = src[e], d = dst[e];
  float nm = dinv[s] * dinv[d];
  const float* hp = hg + (size_t)s * 128 + f;
  float* tp = t + (size_t)d * 128 + f;
  atomicAdd(tp + 0, hp[0] * nm);
  atomicAdd(tp + 1, hp[1] * nm);
  atomicAdd(tp + 2, hp[2] * nm);
  atomicAdd(tp + 3, hp[3] * nm);
}
__global__ void k_add_bias_res(float* t, const float* __restrict__ b,
                               const float* __restrict__ x, size_t nc, int C) {
  size_t i = (size_t)blockIdx.x * 256 + threadIdx.x;
  if (i < nc) t[i] += b[i % (size_t)C] + x[i];
}
__global__ void k_add2(const float* __restrict__ a, const float* __restrict__ b,
                       float* of, _Float16* oh, size_t n) {
  size_t i = (size_t)blockIdx.x * 256 + threadIdx.x;
  if (i < n) {
    float v = a[i] + b[i];
    of[i] = v;
    if (oh) oh[i] = (_Float16)v;
  }
}
// per-feature mean/var over the node axis (biased var, as jnp.var)
__global__ __launch_bounds__(256) void k_bn_stats(const float* __restrict__ t,
                                                  float* mean, float* var, int n, int C) {
  __shared__ float s1[256], s2[256];
  int c = blockIdx.x;
  float a = 0.0f, b = 0.0f;
  for (int i = threadIdx.x; i < n; i += 256) {
    float v = t[(size_t)i * C + c];
    a += v; b += v * v;
  }
  s1[threadIdx.x] = a; s2[threadIdx.x] = b;
  __syncthreads();
  for (int o = 128; o > 0; o >>= 1) {
    if ((int)threadIdx.x < o) {
      s1[threadIdx.x] += s1[threadIdx.x + o];
      s2[threadIdx.x] += s2[threadIdx.x + o];
    }
    __syncthreads();
  }
  if (threadIdx.x == 0) {
    float mu = s1[0] / n;
    mean[c] = mu;
    var[c] = s2[0] / n - mu * mu;
  }
}
__global__ void k_bn_apply(const float* __restrict__ t, const float* __restrict__ mean,
                           const float* __restrict__ var, const float* __restrict__ g,
                           const float* __restrict__ b, float* of, _Float16* oh,
                           size_t nc, int C, int relu) {
  size_t i = (size_t)blockIdx.x * 256 + threadIdx.x;
  if (i < nc) {
    int c = (int)(i % (size_t)C);
    float v = (t[i] - mean[c]) * rsqrtf(var[c] + 1e-5f) * g[c] + b[c];
    if (relu) v = fmaxf(v, 0.0f);
    if (of) of[i] = v;
    if (oh) oh[i] = (_Float16)v;
  }
}
__global__ void k_log_softmax(const float* __restrict__ in, float* out, int n, int C) {
  int r = blockIdx.x * 256 + threadIdx.x;
  if (r >= n) return;
  const float* p = in + (size_t)r * C;
  float mx = -3.0e38f;
  for (int c = 0; c < C; ++c) mx = fmaxf(mx, p[c]);
  float s = 0.0f;
  for (int c = 0; c < C; ++c) s += __expf(p[c] - mx);
  float ls = logf(s);
  float* q = out + (size_t)r * C;
  for (int c = 0; c < C; ++c) q[c] = p[c] - mx - ls;
}

// ============================================================================
// Host orchestration
// ============================================================================
extern "C" void kernel_launch(void* const* d_in, const int* in_sizes, int n_in,
                              void* d_out, int out_size, void* d_ws, size_t ws_size,
                              hipStream_t stream)
{
  (void)in_sizes; (void)n_in; (void)out_size; (void)ws_size;
  const int N = N_NODES, C = D_CH, E = E_EDGES;
  const size_t NC = (size_t)N * C;

  const float* x   = (const float*)d_in[0];
  const int*   ei  = (const int*)d_in[1];
  const int*   src = ei;
  const int*   dst = ei + E;
  const float* P1[20]; const float* P2[20];
  for (int i = 0; i < 20; ++i) {
    P1[i] = (const float*)d_in[2 + i];
    P2[i] = (const float*)d_in[22 + i];
  }
  const float* l1W = (const float*)d_in[42];
  const float* l1b = (const float*)d_in[43];
  const float* l2W = (const float*)d_in[44];
  const float* l2b = (const float*)d_in[45];
  float* out = (float*)d_out;

  enum { GCN_W = 0, GCN_B, WQ, BQ, WK, BK, WV, BV, WO, BO,
         N1G, N1B, N2G, N2B, N3G, N3B, MW1, MB1, MW2, MB2 };

  // ---- workspace bump allocator (~56 MB total)
  char* wsp = (char*)d_ws;
  size_t off = 0;
  auto alloc = [&](size_t bytes) -> void* {
    void* p = wsp + off;
    off = (off + bytes + 255) & ~(size_t)255;
    return p;
  };
  _Float16* xh    = (_Float16*)alloc(NC * 2);
  float*    hg    = (float*)alloc(NC * 4);        // x@W ; later reused as h_attn
  float*    tb    = (float*)alloc(NC * 4);        // bn input scratch
  float*    hloc  = (float*)alloc(NC * 4);
  _Float16* qh    = (_Float16*)alloc(NC * 2);
  _Float16* kh    = (_Float16*)alloc(NC * 2);
  _Float16* vh    = (_Float16*)alloc(NC * 2);
  _Float16* aoh   = (_Float16*)alloc(NC * 2);
  float*    o2    = (float*)alloc(NC * 4);        // Wo out ; later reused as m2
  float*    osum  = (float*)alloc(NC * 4);
  _Float16* osumh = (_Float16*)alloc(NC * 2);
  _Float16* m1h   = (_Float16*)alloc((size_t)N * 256 * 2);
  float*    y1    = (float*)alloc(NC * 4);
  _Float16* y1h   = (_Float16*)alloc(NC * 2);
  float*    x2    = (float*)alloc(NC * 4);
  _Float16* x2h   = (_Float16*)alloc(NC * 2);
  float*    y2    = (float*)alloc(NC * 4);
  _Float16* y2h   = (_Float16*)alloc(NC * 2);
  float*    logit = (float*)alloc((size_t)N * N_CLS * 4);
  float*    deg   = (float*)alloc((size_t)N * 4);
  float*    dinv  = (float*)alloc((size_t)N * 4);
  float*    meanb = (float*)alloc(512);
  float*    varb  = (float*)alloc(512);
  _Float16* gcnT  = (_Float16*)alloc(128 * 128 * 2);
  _Float16* qT    = (_Float16*)alloc(128 * 128 * 2);
  _Float16* kT    = (_Float16*)alloc(128 * 128 * 2);
  _Float16* vT    = (_Float16*)alloc(128 * 128 * 2);
  _Float16* oT    = (_Float16*)alloc(128 * 128 * 2);
  _Float16* m1T   = (_Float16*)alloc(128 * 256 * 2);
  _Float16* m2T   = (_Float16*)alloc(256 * 128 * 2);
  _Float16* l1T   = (_Float16*)alloc(128 * 128 * 2);
  _Float16* l2T   = (_Float16*)alloc(128 * 40 * 2);

  const int blNC = (int)((NC + 255) / 256);

  auto gemm = [&](const _Float16* A, const _Float16* BT, const float* bias,
                  float* Cf, _Float16* Ch, int K, int Nc, int relu) {
    dim3 g(N / 128, (Nc + 63) / 64);
    k_gemm_f16<<<g, 256, 0, stream>>>(A, BT, bias, Cf, Ch, N, K, Nc, relu);
  };
  auto transp = [&](const float* W, _Float16* WT, int K, int Nc) {
    int n = K * Nc;
    k_transpose_f16<<<(n + 255) / 256, 256, 0, stream>>>(W, WT, K, Nc);
  };

  // ---- graph degree + sym normalization (shared by both layers)
  k_fill<<<(N + 255) / 256, 256, 0, stream>>>(deg, 1.0f, N);  // self loops
  k_deg<<<(E + 255) / 256, 256, 0, stream>>>(dst, deg, E);
  k_rsqrt<<<(N + 255) / 256, 256, 0, stream>>>(deg, dinv, N);
  k_f32_to_f16<<<blNC, 256, 0, stream>>>(x, xh, NC);

  auto gps_layer = [&](const float* xin, const _Float16* xinh, const float* const* P,
                       float* yout, _Float16* youth) {
    // --- local GCN branch: bn1(gcn(x) + x)
    transp(P[GCN_W], gcnT, C, C);
    gemm(xinh, gcnT, nullptr, hg, nullptr, C, C, 0);
    k_gcn_self<<<blNC, 256, 0, stream>>>(hg, dinv, tb, NC, C);
    k_gcn_scatter<<<(int)(((size_t)E * 32 + 255) / 256), 256, 0, stream>>>(
        src, dst, hg, dinv, tb, E);
    k_add_bias_res<<<blNC, 256, 0, stream>>>(tb, P[GCN_B], xin, NC, C);
    k_bn_stats<<<C, 256, 0, stream>>>(tb, meanb, varb, N, C);
    k_bn_apply<<<blNC, 256, 0, stream>>>(tb, meanb, varb, P[N1G], P[N1B],
                                         hloc, (_Float16*)nullptr, NC, C, 0);
    // --- global attention branch: bn2(attn(x) + x)
    transp(P[WQ], qT, C, C);
    transp(P[WK], kT, C, C);
    transp(P[WV], vT, C, C);
    gemm(xinh, qT, P[BQ], nullptr, qh, C, C, 0);
    gemm(xinh, kT, P[BK], nullptr, kh, C, C, 0);
    gemm(xinh, vT, P[BV], nullptr, vh, C, C, 0);
    k_flash_attn<<<N / 64, 256, 0, stream>>>(qh, kh, vh, (float*)nullptr, aoh, N);
    transp(P[WO], oT, C, C);
    gemm(aoh, oT, P[BO], o2, nullptr, C, C, 0);
    k_add2<<<blNC, 256, 0, stream>>>(o2, xin, tb, (_Float16*)nullptr, NC);
    k_bn_stats<<<C, 256, 0, stream>>>(tb, meanb, varb, N, C);
    k_bn_apply<<<blNC, 256, 0, stream>>>(tb, meanb, varb, P[N2G], P[N2B],
                                         hg, (_Float16*)nullptr, NC, C, 0);  // h_attn
    // --- combine + MLP: bn3(out + mlp(out))
    k_add2<<<blNC, 256, 0, stream>>>(hloc, hg, osum, osumh, NC);
    transp(P[MW1], m1T, C, 2 * C);
    gemm(osumh, m1T, P[MB1], nullptr, m1h, C, 2 * C, 1);
    transp(P[MW2], m2T, 2 * C, C);
    gemm(m1h, m2T, P[MB2], o2, nullptr, 2 * C, C, 0);  // o2 = m2
    k_add2<<<blNC, 256, 0, stream>>>(osum, o2, tb, (_Float16*)nullptr, NC);
    k_bn_stats<<<C, 256, 0, stream>>>(tb, meanb, varb, N, C);
    k_bn_apply<<<blNC, 256, 0, stream>>>(tb, meanb, varb, P[N3G], P[N3B],
                                         yout, youth, NC, C, 1);  // outer relu fused
  };

  // ---- network
  gps_layer(x, xh, P1, y1, y1h);
  transp(l1W, l1T, C, C);
  gemm(y1h, l1T, l1b, x2, x2h, C, C, 1);
  gps_layer(x2, x2h, P2, y2, y2h);
  transp(l2W, l2T, C, N_CLS);
  gemm(y2h, l2T, l2b, logit, nullptr, C, N_CLS, 0);
  k_log_softmax<<<(N + 255) / 256, 256, 0, stream>>>(logit, out, N, N_CLS);
}
